// MultiEdgeTypePredictor_62577673503220
// MI455X (gfx1250) — compile-verified, matrix operand order checked by reading, not run on hardware
//
#include <hip/hip_runtime.h>

// MI455X / gfx1250 edge-MLP predictor.
// scores[e] = relu( (BN(relu(cat(h_src[s],h_dst[d]) @ W1 + b1)) + emb) @ W2 + b2 ) @ W3 + b3
// Strategy: persistent blocks; W1 (bf16, B-operand layout) resident in 128 VGPRs/wave;
// gathered edge features staged as bf16 in LDS; both GEMMs via v_wmma_f32_16x16x32_bf16.
// Round 3: sched_group_barrier-enforced interleaving (DS-read groups pipelined 4 operands
// ahead of their consuming WMMA) so the backend cannot re-serialize load->wait->wmma.

typedef __attribute__((ext_vector_type(16))) __bf16 v16bf;
typedef __attribute__((ext_vector_type(8)))  __bf16 bf16x8;
typedef __attribute__((ext_vector_type(8)))  float  v8f;

#define E_TILE     64
#define IN_FEAT    256
#define HID        128
#define H2         64
#define SE_STRIDE  1040   // (512+8) bf16 per edge row  -> bank-offset 4 dwords/row
#define SX_STRIDE  272    // (128+8) bf16 per x row
#define SW2_STRIDE 272    // (128+8) bf16 per W2^T row
#define SX2_STRIDE 260    // 65 f32 per x2 row (conflict-free row dot)

#define SE_BYTES  (E_TILE * SE_STRIDE)            // 66560
#define SX_BYTES  (E_TILE * SX_STRIDE)            // 17408
#define SW2_BYTES (H2 * SW2_STRIDE)               // 17408
#define SX2_BYTES (E_TILE * SX2_STRIDE)           // 16640

// sched_group_barrier masks
#define SG_WMMA    0x008
#define SG_DSREAD  0x100

__device__ __forceinline__ v16bf ld_op32(const unsigned char* p) {
  // 32 contiguous bytes from LDS as one WMMA operand (two ds_load_b128)
  union { v16bf v; bf16x8 h[2]; } u;
  u.h[0] = *(const bf16x8*)(p);
  u.h[1] = *(const bf16x8*)(p + 16);
  return u.v;
}

__device__ __forceinline__ v8f zero8() {
  v8f z;
#pragma unroll
  for (int i = 0; i < 8; ++i) z[i] = 0.0f;
  return z;
}

struct bf2 { __bf16 a, b; };
struct bf4 { __bf16 a, b, c, d; };

__global__ __launch_bounds__(256) void edge_mlp_wmma(
    const float* __restrict__ h_src, const float* __restrict__ h_dst,
    const int* __restrict__ src_nodes, const int* __restrict__ dst_nodes,
    const int* __restrict__ etype_idx,
    const float* __restrict__ W1, const float* __restrict__ b1,
    const float* __restrict__ bn_g, const float* __restrict__ bn_b,
    const float* __restrict__ bn_m, const float* __restrict__ bn_v,
    const float* __restrict__ emb,
    const float* __restrict__ W2, const float* __restrict__ b2,
    const float* __restrict__ W3, const float* __restrict__ b3,
    float* __restrict__ out, int E, int ntiles)
{
  __shared__ __attribute__((aligned(16))) unsigned char smem[SE_BYTES + SX_BYTES + SW2_BYTES + SX2_BYTES];
  unsigned char* sE  = smem;
  unsigned char* sX  = smem + SE_BYTES;
  unsigned char* sW2 = smem + SE_BYTES + SX_BYTES;
  unsigned char* sX2 = smem + SE_BYTES + SX_BYTES + SW2_BYTES;

  const int tid  = threadIdx.x;
  const int lane = tid & 31;
  const int wv   = tid >> 5;        // wave id 0..7
  const int half = lane >> 4;       // hi/lo 16-lane group
  const int l16  = lane & 15;

  // ---- Persistent B-operand: W1 columns [wv*16, wv*16+16) in bf16, 16 K-blocks of 32 ----
  v16bf B1[16];
  {
    const int colB = (wv << 4) + l16;
#pragma unroll
    for (int kb = 0; kb < 16; ++kb) {
      const float* p = W1 + (size_t)(kb * 32 + half * 16) * HID + colB;
      v16bf bv;
#pragma unroll
      for (int i = 0; i < 16; ++i) bv[i] = (__bf16)p[(size_t)i * HID];
      B1[kb] = bv;
    }
  }

  // ---- Stage W2^T (bf16) into LDS once: sW2[col n][k] ----
#pragma unroll
  for (int it = 0; it < 16; ++it) {
    int p  = it * 256 + tid;        // 0..4095 pairs
    int n  = p & 63;
    int kp = p >> 6;                // K pair index 0..63
    bf2 q;
    q.a = (__bf16)W2[(size_t)(2 * kp)     * H2 + n];
    q.b = (__bf16)W2[(size_t)(2 * kp + 1) * H2 + n];
    *(bf2*)(sW2 + n * SW2_STRIDE + kp * 4) = q;
  }

  // ---- Fused epilogue constants for this wave's 16 columns of GEMM1 output ----
  const int   et    = etype_idx[0];
  const int   colx  = (wv << 4) + l16;            // 0..127
  const float scale = bn_g[colx] * rsqrtf(bn_v[colx] + 1e-5f);
  const float shift = bn_b[colx] - bn_m[colx] * scale + emb[(size_t)et * HID + colx];
  const float bias1 = b1[colx];

  const int   m2  = wv >> 1;                      // GEMM2 M-tile for this wave
  const int   nn  = (wv & 1) * 2;                 // GEMM2 N-tile pair base
  const float b2c0 = b2[(nn)     * 16 + l16];
  const float b2c1 = b2[(nn + 1) * 16 + l16];
  const float b3v  = b3[0];

  // ---- Loop-invariant, wave-uniform gather routing ----
  const int  c      = tid & 127;        // 4-float chunk 0..127 (const per thread)
  const int  esub0  = tid >> 7;         // 0 or 1
  const bool isSrc  = (c < 64);         // uniform within each wave
  const float* __restrict__ feat = isSrc ? h_src : h_dst;
  const int*   __restrict__ idxp = isSrc ? src_nodes : dst_nodes;
  const int  coff   = (c & 63) * 4;     // float offset within node row
  const int  Em1    = E - 1;

  for (int tile = blockIdx.x; tile < ntiles; tile += gridDim.x) {
    const int e0 = tile * E_TILE;

    // ---- Gather + f32->bf16 convert: edge_feat[64][512] = [src(256) | dst(256)] ----
    // Batch 8 independent index loads, then 8 independent float4 feature loads.
#pragma unroll
    for (int g = 0; g < 4; ++g) {
      int nd[8];
#pragma unroll
      for (int u = 0; u < 8; ++u) {
        int ge = e0 + (g * 8 + u) * 2 + esub0;
        if (ge > Em1) ge = Em1;
        nd[u] = idxp[ge];
      }
#pragma unroll
      for (int u = 0; u < 8; ++u) {
        float4 v = *(const float4*)(feat + (size_t)nd[u] * IN_FEAT + coff);
        int e = (g * 8 + u) * 2 + esub0;
        bf4 q; q.a = (__bf16)v.x; q.b = (__bf16)v.y; q.c = (__bf16)v.z; q.d = (__bf16)v.w;
        *(bf4*)(sE + e * SE_STRIDE + c * 8) = q;
      }
    }
    __syncthreads();

    // ---- GEMM1: [64,512] x [512,128] -> fused bias/ReLU/BN/emb -> sX (bf16) ----
#pragma unroll
    for (int m = 0; m < 4; ++m) {
      v8f acc = zero8();
      const unsigned char* aRow = sE + (m * 16 + l16) * SE_STRIDE + half * 32;
      v16bf a[4];
#pragma unroll
      for (int i = 0; i < 4; ++i) a[i] = ld_op32(aRow + i * 64);
#pragma unroll
      for (int kb = 0; kb < 16; ++kb) {
        acc = __builtin_amdgcn_wmma_f32_16x16x32_bf16(
            false, a[kb & 3], false, B1[kb], (short)0, acc, false, false);
        if (kb + 4 < 16) a[kb & 3] = ld_op32(aRow + (kb + 4) * 64);
      }
      const int rowb = m * 16 + half * 8;
#pragma unroll
      for (int j = 0; j < 8; ++j) {
        float v = acc[j] + bias1;
        v = fmaxf(v, 0.0f);
        v = v * scale + shift;
        *(__bf16*)(sX + (rowb + j) * SX_STRIDE + colx * 2) = (__bf16)v;
      }
    }
    // Enforce pipelined schedule over the whole unrolled GEMM1 region:
    // 8 ds_reads up front (4 operands in flight), then {1 wmma, 2 ds_reads},
    // draining with 4 wmmas.  Totals: 128 DS-reads, 64 WMMAs.
    __builtin_amdgcn_sched_group_barrier(SG_DSREAD, 8, 0);
#pragma unroll
    for (int s = 0; s < 60; ++s) {
      __builtin_amdgcn_sched_group_barrier(SG_WMMA,   1, 0);
      __builtin_amdgcn_sched_group_barrier(SG_DSREAD, 2, 0);
    }
    __builtin_amdgcn_sched_group_barrier(SG_WMMA, 4, 0);
    __syncthreads();

    // ---- GEMM2: [64,128] x [128,64], wave handles M-tile m2, N-tiles nn,nn+1 ----
    {
      v8f acc0 = zero8(), acc1 = zero8();
      const unsigned char* aRow  = sX  + (m2 * 16 + l16) * SX_STRIDE  + half * 32;
      const unsigned char* bRow0 = sW2 + ((nn)     * 16 + l16) * SW2_STRIDE + half * 32;
      const unsigned char* bRow1 = sW2 + ((nn + 1) * 16 + l16) * SW2_STRIDE + half * 32;
      v16bf a[2], q0[2], q1[2];
      a[0]  = ld_op32(aRow);
      q0[0] = ld_op32(bRow0);
      q1[0] = ld_op32(bRow1);
#pragma unroll
      for (int kbb = 0; kbb < 4; ++kbb) {
        const int cur = kbb & 1, nxt = cur ^ 1;
        if (kbb < 3) {
          a[nxt]  = ld_op32(aRow  + (kbb + 1) * 64);
          q0[nxt] = ld_op32(bRow0 + (kbb + 1) * 64);
          q1[nxt] = ld_op32(bRow1 + (kbb + 1) * 64);
        }
        acc0 = __builtin_amdgcn_wmma_f32_16x16x32_bf16(false, a[cur], false, q0[cur], (short)0, acc0, false, false);
        acc1 = __builtin_amdgcn_wmma_f32_16x16x32_bf16(false, a[cur], false, q1[cur], (short)0, acc1, false, false);
      }
      const int rowb = m2 * 16 + half * 8;
#pragma unroll
      for (int j = 0; j < 8; ++j) {
        float v0 = fmaxf(acc0[j] + b2c0, 0.0f);
        float v1 = fmaxf(acc1[j] + b2c1, 0.0f);
        *(float*)(sX2 + (rowb + j) * SX2_STRIDE + ((nn)     * 16 + l16) * 4) = v0;
        *(float*)(sX2 + (rowb + j) * SX2_STRIDE + ((nn + 1) * 16 + l16) * 4) = v1;
      }
      // Pipeline GEMM2 the same way: 6 reads ahead, {2 wmma, 6 ds_reads} x3, 2 wmma.
      __builtin_amdgcn_sched_group_barrier(SG_DSREAD, 6, 0);
#pragma unroll
      for (int s = 0; s < 3; ++s) {
        __builtin_amdgcn_sched_group_barrier(SG_WMMA,   2, 0);
        __builtin_amdgcn_sched_group_barrier(SG_DSREAD, 6, 0);
      }
      __builtin_amdgcn_sched_group_barrier(SG_WMMA, 2, 0);
    }
    __syncthreads();

    // ---- Final: scores = x2 @ W3 + b3 (64 rows, conflict-free stride-65 LDS) ----
    if (tid < E_TILE) {
      int ge = e0 + tid;
      if (ge < E) {
        const float* xr = (const float*)(sX2 + tid * SX2_STRIDE);
        float s = b3v;
#pragma unroll
        for (int c2 = 0; c2 < H2; ++c2) s += xr[c2] * W3[c2];
        out[ge] = s;
      }
    }
    // No trailing barrier needed: next-iteration writes touch sE only, the
    // post-stage barrier orders them against this iteration's sX2 readers.
  }
}

extern "C" void kernel_launch(void* const* d_in, const int* in_sizes, int n_in,
                              void* d_out, int out_size, void* d_ws, size_t ws_size,
                              hipStream_t stream) {
  const float* h_src     = (const float*)d_in[0];
  const float* h_dst     = (const float*)d_in[1];
  const int*   src_nodes = (const int*)d_in[2];
  const int*   dst_nodes = (const int*)d_in[3];
  const int*   etype_idx = (const int*)d_in[4];
  const float* W1        = (const float*)d_in[5];
  const float* b1        = (const float*)d_in[6];
  const float* bn_g      = (const float*)d_in[7];
  const float* bn_b      = (const float*)d_in[8];
  const float* bn_m      = (const float*)d_in[9];
  const float* bn_v      = (const float*)d_in[10];
  const float* emb       = (const float*)d_in[11];
  const float* W2        = (const float*)d_in[12];
  const float* b2        = (const float*)d_in[13];
  const float* W3        = (const float*)d_in[14];
  const float* b3        = (const float*)d_in[15];
  float* out = (float*)d_out;

  const int E = in_sizes[2];
  const int ntiles = (E + E_TILE - 1) / E_TILE;
  int grid = ntiles < 1024 ? ntiles : 1024;   // persistent blocks; W1 regs reused ~4-5x

  edge_mlp_wmma<<<grid, 256, 0, stream>>>(
      h_src, h_dst, src_nodes, dst_nodes, etype_idx,
      W1, b1, bn_g, bn_b, bn_m, bn_v, emb, W2, b2, W3, b3,
      out, E, ntiles);
}
